// GraphMatLayers_23914377904731
// MI455X (gfx1250) — matrix operand order repaired and checked by reading, not compiled, not run on hardware
//
#include <hip/hip_runtime.h>

// GraphMatLayers for MI455X (gfx1250): N=256,M=128,C=64,P=128,GS=4,L=8.
// One workgroup (256 thr = 8 wave32) runs the full 8-layer net for one sample.
// bf16 WMMA 16x16x32 with fp32 accumulation; all operands staged in LDS.

typedef __bf16 bf16_t;
typedef bf16_t v16bf __attribute__((ext_vector_type(16)));
typedef bf16_t v8bf  __attribute__((ext_vector_type(8)));
typedef bf16_t v2bf  __attribute__((ext_vector_type(2)));
typedef float  v8f   __attribute__((ext_vector_type(8)));
typedef unsigned int v4u __attribute__((ext_vector_type(4)));

#define LDH 136   // padded LDS row stride (bf16): 68 dwords -> bank step 4, conflict-free

// two floats -> packed bf16x2; fptrunc is RNE, backend picks the native cvt if present
__device__ __forceinline__ unsigned int pk2bf(float a, float b) {
  v2bf v = { (bf16_t)a, (bf16_t)b };
  return __builtin_bit_cast(unsigned int, v);
}
__device__ __forceinline__ float bf2f(unsigned int u) { return __uint_as_float(u << 16); }

__device__ __forceinline__ v8f wmma_bf16(v16bf a, v16bf b, v8f c) {
  return __builtin_amdgcn_wmma_f32_16x16x32_bf16(false, a, false, b, (short)0, c, false, false);
}

// A-matrix 16x32 bf16 frag (ISA 7.12.2): lane r=L&15 -> row M=r; group h=L>>4:
// K = k0 + h*8 + [0..7]  and  K = k0 + 16 + h*8 + [0..7]  -> two contiguous 16B LDS loads.
__device__ __forceinline__ v16bf load_a_frag(const unsigned short* base, int row0, int k0) {
  int lane = threadIdx.x & 31;
  const unsigned short* p = base + (row0 + (lane & 15)) * LDH + k0 + ((lane >> 4) << 3);
  v8bf lo = *(const v8bf*)p;
  v8bf hi = *(const v8bf*)(p + 16);
  return __builtin_shufflevector(lo, hi, 0,1,2,3,4,5,6,7,8,9,10,11,12,13,14,15);
}
// B-matrix 32x16 bf16 frag: lane r -> col N=r; group h: K = k0 + h*8 + [0..7] and K = k0 + 16 + h*8 + [0..7].
__device__ __forceinline__ v16bf load_b_frag(const unsigned short* base, int n0, int k0) {
  int lane = threadIdx.x & 31;
  const unsigned short* p = base + (n0 + (lane & 15)) * LDH + k0 + ((lane >> 4) << 3);
  v8bf lo = *(const v8bf*)p;
  v8bf hi = *(const v8bf*)(p + 16);
  return __builtin_shufflevector(lo, hi, 0,1,2,3,4,5,6,7,8,9,10,11,12,13,14,15);
}

// fp32 global [128][2^colsLog2] -> bf16 LDS [128][LDH], coalesced float4 loads
__device__ __forceinline__ void stage_bf16(unsigned short* dst, const float* __restrict__ src,
                                           int colsLog2) {
  const int total = 128 << colsLog2;
  const int mask  = (1 << colsLog2) - 1;
  for (int e = (int)threadIdx.x * 4; e < total; e += 1024) {
    const float4 v = *(const float4*)(src + e);
    const int r = e >> colsLog2, c = e & mask;
    *(uint2*)(dst + r * LDH + c) = make_uint2(pk2bf(v.x, v.y), pk2bf(v.z, v.w));
  }
}

// One GNN layer: h_next = [RES ? h : 0] + sum_g relu(G_g @ (h @ W_g^T + b_g))
// KD = inner dim of step A (64 for layer 0, 128 otherwise). Constant -> full unroll.
template <int KD, bool RES, bool LAST>
__device__ __forceinline__ void layer_body(
    unsigned short* sh_h, unsigned short* sh_w, unsigned short* sh_y,
    unsigned short* sh_g, float* sh_b,
    const float* __restrict__ Wbase, const float* __restrict__ bbase,
    const float* __restrict__ Gbase, float* __restrict__ outp)
{
  const int wave  = threadIdx.x >> 5;
  const int lane  = threadIdx.x & 31;
  const int col   = lane & 15;            // N-column inside a C/D tile
  const int rbase = (lane >> 4) << 3;     // C/D row offset (0 or 8)
  constexpr int WCL = (KD == 64) ? 6 : 7;

  v8f zsum[8];
  #pragma unroll
  for (int t = 0; t < 8; ++t)
    #pragma unroll
    for (int j = 0; j < 8; ++j) zsum[t][j] = 0.0f;

  #pragma unroll 1
  for (int g = 0; g < 4; ++g) {
    __syncthreads();   // prev step-B readers done before restaging
    stage_bf16(sh_w, Wbase + (size_t)g * 128 * KD, WCL);
    stage_bf16(sh_g, Gbase + (size_t)g * 128 * 128, 7);
    if (threadIdx.x < 128) sh_b[threadIdx.x] = bbase[g * 128 + threadIdx.x];
    __syncthreads();

    // ---- Step A: y[m][p] = sum_c h[m][c] * W[p][c]; wave owns rows m=[16w,16w+16)
    v8f yacc[8];
    #pragma unroll
    for (int t = 0; t < 8; ++t)
      #pragma unroll
      for (int j = 0; j < 8; ++j) yacc[t][j] = 0.0f;

    #pragma unroll
    for (int ks = 0; ks < KD; ks += 32) {
      v16bf a = load_a_frag(sh_h, wave * 16, ks);
      #pragma unroll
      for (int tp = 0; tp < 8; ++tp) {
        v16bf b = load_b_frag(sh_w, tp * 16, ks);
        yacc[tp] = wmma_bf16(a, b, yacc[tp]);
      }
    }
    // epilogue: +bias, pack bf16, store transposed yT[p][m] (contiguous along m)
    #pragma unroll
    for (int tp = 0; tp < 8; ++tp) {
      const int p = tp * 16 + col;
      const float bb = sh_b[p];
      v4u pk;
      #pragma unroll
      for (int j = 0; j < 4; ++j)
        pk[j] = pk2bf(yacc[tp][2 * j] + bb, yacc[tp][2 * j + 1] + bb);
      *(v4u*)(sh_y + p * LDH + wave * 16 + rbase) = pk;
    }
    __syncthreads();

    // ---- Step B: zT[p][i] = sum_k yT[p][k] * G[i][k]; wave owns rows p=[16w,16w+16)
    v8f zg[8];
    #pragma unroll
    for (int t = 0; t < 8; ++t)
      #pragma unroll
      for (int j = 0; j < 8; ++j) zg[t][j] = 0.0f;

    #pragma unroll
    for (int ks = 0; ks < 128; ks += 32) {
      v16bf a = load_a_frag(sh_y, wave * 16, ks);
      #pragma unroll
      for (int ti = 0; ti < 8; ++ti) {
        v16bf b = load_b_frag(sh_g, ti * 16, ks);
        zg[ti] = wmma_bf16(a, b, zg[ti]);
      }
    }
    #pragma unroll
    for (int ti = 0; ti < 8; ++ti)
      #pragma unroll
      for (int j = 0; j < 8; ++j)
        zsum[ti][j] += fmaxf(zg[ti][j], 0.0f);   // ReLU per channel, sum over g
  } // g

  // zsum tile ti element: (i = ti*16+col, p = 16*wave + rbase + j)
  const int pbase = wave * 16 + rbase;
  if (RES) {   // residual: h_next += h (h kept in bf16 between layers)
    #pragma unroll
    for (int ti = 0; ti < 8; ++ti) {
      v4u hv = *(const v4u*)(sh_h + (ti * 16 + col) * LDH + pbase);
      #pragma unroll
      for (int j = 0; j < 4; ++j) {
        zsum[ti][2 * j]     += bf2f(hv[j] & 0xFFFFu);
        zsum[ti][2 * j + 1] += bf2f(hv[j] >> 16);
      }
    }
  }
  if (LAST) {  // final fp32 output [i][p], 32B-aligned contiguous per lane
    #pragma unroll
    for (int ti = 0; ti < 8; ++ti) {
      float* oo = outp + (ti * 16 + col) * 128 + pbase;
      *(v8f*)oo = zsum[ti];
    }
  } else {
    __syncthreads();   // all reads of sh_h complete before overwrite
    #pragma unroll
    for (int ti = 0; ti < 8; ++ti) {
      v4u pk;
      #pragma unroll
      for (int j = 0; j < 4; ++j)
        pk[j] = pk2bf(zsum[ti][2 * j], zsum[ti][2 * j + 1]);
      *(v4u*)(sh_h + (ti * 16 + col) * LDH + pbase) = pk;
    }
    // next g-loop's leading __syncthreads orders these writes before step-A reads
  }
}

__global__ __launch_bounds__(256)
void gml_kernel(const float* __restrict__ G,  const float* __restrict__ x,
                const float* __restrict__ W0, const float* __restrict__ b0,
                const float* __restrict__ Ws, const float* __restrict__ bs,
                float* __restrict__ out)
{
  extern __shared__ unsigned char smem[];
  unsigned short* sh_h = (unsigned short*)smem;   // h  [m][p] (layer0 [m][c])
  unsigned short* sh_w = sh_h + 128 * LDH;        // W  [p][c]
  unsigned short* sh_y = sh_w + 128 * LDH;        // yT [p][m]
  unsigned short* sh_g = sh_y + 128 * LDH;        // Gg [i][k]
  float*          sh_b = (float*)(sh_g + 128 * LDH);

  const int n = blockIdx.x;
  const float* Gn = G + (size_t)n * 4 * 128 * 128;

  stage_bf16(sh_h, x + (size_t)n * 128 * 64, 6);
  __syncthreads();

  // layer 0: C=64, no residual
  layer_body<64, false, false>(sh_h, sh_w, sh_y, sh_g, sh_b, W0, b0, Gn, nullptr);

  // layers 1..6: K=128, residual, stay in LDS
  #pragma unroll 1
  for (int l = 1; l < 7; ++l)
    layer_body<128, true, false>(sh_h, sh_w, sh_y, sh_g, sh_b,
                                 Ws + (size_t)(l - 1) * 4 * 128 * 128,
                                 bs + (size_t)(l - 1) * 4 * 128, Gn, nullptr);

  // layer 7: residual, write fp32 result
  layer_body<128, true, true>(sh_h, sh_w, sh_y, sh_g, sh_b,
                              Ws + (size_t)6 * 4 * 128 * 128,
                              bs + (size_t)6 * 4 * 128, Gn,
                              out + (size_t)n * 128 * 128);
}

extern "C" void kernel_launch(void* const* d_in, const int* in_sizes, int n_in,
                              void* d_out, int out_size, void* d_ws, size_t ws_size,
                              hipStream_t stream) {
  const float* G  = (const float*)d_in[0];
  const float* x  = (const float*)d_in[1];
  const float* W0 = (const float*)d_in[2];
  const float* b0 = (const float*)d_in[3];
  const float* Ws = (const float*)d_in[4];
  const float* bs = (const float*)d_in[5];
  float* out = (float*)d_out;

  const size_t smem = (size_t)4 * 128 * LDH * sizeof(unsigned short) + 128 * sizeof(float);
  hipFuncSetAttribute((const void*)gml_kernel,
                      hipFuncAttributeMaxDynamicSharedMemorySize, (int)smem);
  gml_kernel<<<256, 256, smem, stream>>>(G, x, W0, b0, Ws, bs, out);
}
